// SelfAttention_15599321219155
// MI455X (gfx1250) — matrix operand order repaired and checked by reading, not compiled, not run on hardware
//
#include <hip/hip_runtime.h>

// Problem sizes (fixed by the reference)
#define BB   4
#define SS   4096
#define DM   256
#define DK   64
#define ROWS (BB * SS)   // 16384 flattened (b,s) rows
#define NKT  (SS / 32)   // 128 key tiles of 32

typedef __attribute__((ext_vector_type(2)))  float        v2f;
typedef __attribute__((ext_vector_type(8)))  float        v8f;
typedef __attribute__((ext_vector_type(8)))  _Float16     v8h;
typedef __attribute__((ext_vector_type(16))) _Float16     v16h;
typedef __attribute__((ext_vector_type(4)))  unsigned int v4u;
typedef __attribute__((ext_vector_type(8)))  int          v8i;
typedef __attribute__((ext_vector_type(4)))  int          v4i;

// ---- WMMA wrappers (CDNA5 builtins, 8-arg form) ----------------------------
__device__ __forceinline__ v8f wmma_f16(v16h a, v16h b, v8f c) {
  return __builtin_amdgcn_wmma_f32_16x16x32_f16(false, a, false, b,
                                                (short)0, c, false, false);
}
__device__ __forceinline__ v8f wmma_f32k4(v2f a, v2f b, v8f c) {
  return __builtin_amdgcn_wmma_f32_16x16x4_f32(false, a, false, b,
                                               (short)0, c, false, false);
}

// Build a 16x32 A-fragment: elements 0..7 <- lo[0..7], 8..15 <- hi[0..7]
__device__ __forceinline__ v16h make_a(const _Float16* lo, const _Float16* hi) {
  v8h l = *(const v8h*)lo;
  v8h h = *(const v8h*)hi;
  v16h r;
#pragma unroll
  for (int i = 0; i < 8; ++i) { r[i] = l[i]; r[i + 8] = h[i]; }
  return r;
}

// ---- Tensor Data Mover: D# descriptor builders (ISA ch.8, §8.3-8.6) -------
// Group 0: [1:0]=count(1) | lds_addr @[63:32] | global_addr @[120:64] | type=2
__device__ __forceinline__ v4u tdm_g0(unsigned lds_addr,
                                      unsigned long long gaddr) {
  v4u g0;
  g0[0] = 1u;                                           // count=1, user desc
  g0[1] = lds_addr;                                     // LDS byte address
  g0[2] = (unsigned)(gaddr & 0xFFFFFFFFu);              // global_addr[31:0]
  g0[3] = (unsigned)((gaddr >> 32) & 0x01FFFFFFu)       // global_addr[56:32]
        | (2u << 30);                                   // type = 2
  return g0;
}
// Group 1: 2-byte elements, 2D tile: tensor/tile dims + dim0 stride.
//  [17:16]=data_size(1=2B)  [79:48]=tensor_dim0  [111:80]=tensor_dim1
//  [127:112]=tile_dim0 [143:128]=tile_dim1 [207:160]=tensor_dim0_stride
__device__ __forceinline__ v8i tdm_g1(unsigned td0, unsigned td1,
                                      unsigned tl0, unsigned tl1,
                                      unsigned long long stride0) {
  v8i g1;
  g1[0] = (int)(1u << 16);                              // data_size = 2 bytes
  g1[1] = (int)((td0 & 0xFFFFu) << 16);
  g1[2] = (int)(((td0 >> 16) & 0xFFFFu) | ((td1 & 0xFFFFu) << 16));
  g1[3] = (int)(((td1 >> 16) & 0xFFFFu) | ((tl0 & 0xFFFFu) << 16));
  g1[4] = (int)(tl1 & 0xFFFFu);                         // tile_dim2 = 0
  g1[5] = (int)(unsigned)(stride0 & 0xFFFFFFFFu);
  g1[6] = (int)(unsigned)((stride0 >> 32) & 0xFFFFu);   // dim1_stride = 0
  g1[7] = 0;
  return g1;
}
__device__ __forceinline__ void tdm_load(unsigned lds, unsigned long long ga,
                                         v8i g1) {
  v4i z = {};
#if __clang_major__ >= 23
  v8i z8 = {};
  __builtin_amdgcn_tensor_load_to_lds(tdm_g0(lds, ga), g1, z, z, z8, 0);
#else
  __builtin_amdgcn_tensor_load_to_lds(tdm_g0(lds, ga), g1, z, z, 0);
#endif
}

// ---------------------------------------------------------------------------
// Kernel 1: fused Q/K projection + ReLU, f32 WMMA (16x16x4), emit f16.
// ---------------------------------------------------------------------------
__global__ __launch_bounds__(128) void proj_relu_f16(
    const float* __restrict__ q, const float* __restrict__ k,
    const float* __restrict__ wq, const float* __restrict__ wk,
    _Float16* __restrict__ qp, _Float16* __restrict__ kp) {
  const int tilesPerTensor = (ROWS / 16) * (DK / 16);  // 4096
  int wave = blockIdx.x * (blockDim.x >> 5) + (threadIdx.x >> 5);
  int lane = threadIdx.x & 31;
  int h = lane >> 4, mn = lane & 15;

  const float* src; const float* wsrc; _Float16* dst; int t = wave;
  if (t < tilesPerTensor) { src = q; wsrc = wq; dst = qp; }
  else                    { src = k; wsrc = wk; dst = kp; t -= tilesPerTensor; }

  int row0 = (t >> 2) * 16;
  int col0 = (t & 3) * 16;

  const float* arow = src  + (size_t)(row0 + mn) * DM + 2 * h;
  const float* brow = wsrc + (size_t)(col0 + mn) * DM + 2 * h;

  v8f acc = {};
#pragma unroll 8
  for (int kk = 0; kk < DM; kk += 4) {
    v2f a = *(const v2f*)(arow + kk);
    v2f b = *(const v2f*)(brow + kk);
    acc = wmma_f32k4(a, b, acc);
  }
#pragma unroll
  for (int j = 0; j < 8; ++j) {
    float x = acc[j];
    x = x > 0.0f ? x : 0.0f;
    dst[(size_t)(row0 + j + 8 * h) * DK + col0 + mn] = (_Float16)x;
  }
}

// ---------------------------------------------------------------------------
// Kernel 2: V -> f16 transposed to [b][n][s]; coalesced writes.
// ---------------------------------------------------------------------------
__global__ __launch_bounds__(256) void v_to_f16T(const float* __restrict__ v,
                                                 _Float16* __restrict__ vT) {
  int idx = blockIdx.x * blockDim.x + threadIdx.x;  // (b,n,s), s fastest
  int s  = idx & (SS - 1);
  int bn = idx >> 12;
  int n  = bn & 63;
  int b  = bn >> 6;
  vT[idx] = (_Float16)v[((size_t)(b * SS + s)) * DK + n];
}

// ---------------------------------------------------------------------------
// Kernel 3: flash attention, TDM double-buffered K/V staging.
// Block = 4 waves = 4 adjacent query tiles of one batch -> shared key stream.
// Wave 0 drives the Tensor Data Mover (K tile 32x64, V^T tile 64x32 per step),
// s_wait_tensorcnt + barriers hand off buffers; B-fragments come from LDS.
// ---------------------------------------------------------------------------
__global__ __launch_bounds__(128) void flash_attn(
    const _Float16* __restrict__ qp, const _Float16* __restrict__ kp,
    const _Float16* __restrict__ vT, float* __restrict__ out) {
  __shared__ __align__(32) _Float16 kbuf[2][32 * 64];   // [key][dk]
  __shared__ __align__(32) _Float16 vbuf[2][64 * 32];   // [n][key]
  __shared__ __align__(32) _Float16 pbuf[4][16][32];

  int wid  = threadIdx.x >> 5;
  int lane = threadIdx.x & 31;
  int h = lane >> 4, mn = lane & 15;

  int T  = blockIdx.x * 4 + wid;      // query tile id, 0..1023
  int g0 = T * 16;                    // flat (b*S + s) base row
  int bu = (int)(blockIdx.x >> 6);    // batch (uniform: 64 blocks per batch)

  // Resident Q A-fragments (d_k 0..31 and 32..63)
  const _Float16* qrow = qp + (size_t)(g0 + mn) * DK;
  v16h aq0 = make_a(qrow + 16 * h, qrow + 8 + 16 * h);
  v16h aq1 = make_a(qrow + 32 + 16 * h, qrow + 40 + 16 * h);

  v8f o0 = {}, o1 = {}, o2 = {}, o3 = {};
  float mrow[8], lrow[8];
#pragma unroll
  for (int j = 0; j < 8; ++j) { mrow[j] = -3.0e38f; lrow[j] = 0.0f; }

  const _Float16* kbase = kp + (size_t)bu * SS * DK;    // uniform
  const _Float16* vbase = vT + (size_t)bu * DK * SS;    // uniform

  // TDM group-1 descriptors (uniform, loop-invariant)
  v8i g1k = tdm_g1(/*tensor*/ DK, 32, /*tile*/ DK, 32, /*stride0*/ DK);
  v8i g1v = tdm_g1(/*tensor*/ 32, DK, /*tile*/ 32, DK, /*stride0*/ SS);

  auto issue = [&](int tile, int buf) {
    tdm_load((unsigned)(size_t)&kbuf[buf][0],
             (unsigned long long)(size_t)(kbase + (size_t)tile * 32 * DK), g1k);
    tdm_load((unsigned)(size_t)&vbuf[buf][0],
             (unsigned long long)(size_t)(vbase + (size_t)tile * 32), g1v);
  };

  if (wid == 0) { issue(0, 0); issue(1, 1); }           // prologue: 2 tiles

  for (int i = 0; i < NKT; ++i) {
    int buf = i & 1;
    if (wid == 0) {                    // oldest tile (2 TDM ops) must be done
      if (i < NKT - 1) __builtin_amdgcn_s_wait_tensorcnt(2);
      else             __builtin_amdgcn_s_wait_tensorcnt(0);
    }
    __syncthreads();                   // publish buf to all 4 waves

    // ---- scores: two 16x16 tiles (keys mn and 16+mn of this 32-key tile) --
    const _Float16* kr0 = &kbuf[buf][0] + mn * DK;
    const _Float16* kr1 = kr0 + 16 * DK;
    v16h b00 = *(const v16h*)(kr0 + 16 * h);            // d_k  0..31
    v16h b01 = *(const v16h*)(kr0 + 32 + 16 * h);       // d_k 32..63
    v16h b10 = *(const v16h*)(kr1 + 16 * h);
    v16h b11 = *(const v16h*)(kr1 + 32 + 16 * h);

    v8f s0 = {}, s1 = {};
    s0 = wmma_f16(aq0, b00, s0);
    s0 = wmma_f16(aq1, b01, s0);
    s1 = wmma_f16(aq0, b10, s1);
    s1 = wmma_f16(aq1, b11, s1);

#pragma unroll
    for (int j = 0; j < 8; ++j) { s0[j] *= 0.125f; s1[j] *= 0.125f; }
    if (i == 0 && mn == 0) {           // mask attn[:, :, 0] = -1e38
#pragma unroll
      for (int j = 0; j < 8; ++j) s0[j] = -1.0e38f;
    }

    // ---- online softmax; row j+8h lives in one half-wave ----
#pragma unroll
    for (int j = 0; j < 8; ++j) {
      float mx = fmaxf(s0[j], s1[j]);
      mx = fmaxf(mx, __shfl_xor(mx, 1, 32));
      mx = fmaxf(mx, __shfl_xor(mx, 2, 32));
      mx = fmaxf(mx, __shfl_xor(mx, 4, 32));
      mx = fmaxf(mx, __shfl_xor(mx, 8, 32));
      float mnew  = fmaxf(mrow[j], mx);
      float alpha = __expf(mrow[j] - mnew);
      float p0 = __expf(s0[j] - mnew);
      float p1 = __expf(s1[j] - mnew);
      float rs = p0 + p1;
      rs += __shfl_xor(rs, 1, 32);
      rs += __shfl_xor(rs, 2, 32);
      rs += __shfl_xor(rs, 4, 32);
      rs += __shfl_xor(rs, 8, 32);
      lrow[j] = lrow[j] * alpha + rs;
      mrow[j] = mnew;
      o0[j] *= alpha; o1[j] *= alpha; o2[j] *= alpha; o3[j] *= alpha;
      pbuf[wid][j + 8 * h][mn]      = (_Float16)p0;     // C/D -> LDS
      pbuf[wid][j + 8 * h][16 + mn] = (_Float16)p1;
    }

    // wave-private LDS bounce; CDNA5 split counter wait before re-load
    asm volatile("s_wait_dscnt 0x0" ::: "memory");

    const _Float16* prow = &pbuf[wid][mn][0];
    v16h ap = make_a(prow + 16 * h, prow + 8 + 16 * h);

    // ---- P@V: 4 output column tiles, B from LDS V^T tile (keys contig) ----
    const _Float16* vr = &vbuf[buf][0] + mn * 32 + 16 * h;
    v16h bv0 = *(const v16h*)(vr);
    v16h bv1 = *(const v16h*)(vr + 16 * 32);
    v16h bv2 = *(const v16h*)(vr + 32 * 32);
    v16h bv3 = *(const v16h*)(vr + 48 * 32);
    o0 = wmma_f16(ap, bv0, o0);
    o1 = wmma_f16(ap, bv1, o1);
    o2 = wmma_f16(ap, bv2, o2);
    o3 = wmma_f16(ap, bv3, o3);

    __syncthreads();                   // all waves done reading buf
    if (wid == 0 && i + 2 < NKT) issue(i + 2, buf);
  }

  // ---- epilogue: out = O / l ----
#pragma unroll
  for (int j = 0; j < 8; ++j) {
    float inv = 1.0f / lrow[j];
    size_t r = (size_t)(g0 + j + 8 * h) * DK;
    out[r +  0 + mn] = o0[j] * inv;
    out[r + 16 + mn] = o1[j] * inv;
    out[r + 32 + mn] = o2[j] * inv;
    out[r + 48 + mn] = o3[j] * inv;
  }
}

// ---------------------------------------------------------------------------
extern "C" void kernel_launch(void* const* d_in, const int* in_sizes, int n_in,
                              void* d_out, int out_size, void* d_ws,
                              size_t ws_size, hipStream_t stream) {
  const float* q  = (const float*)d_in[0];
  const float* k  = (const float*)d_in[1];
  const float* v  = (const float*)d_in[2];
  const float* wq = (const float*)d_in[3];
  const float* wk = (const float*)d_in[4];
  float* out = (float*)d_out;

  // Workspace: qp(2MB) | kp(2MB) | vT(2MB), all f16
  _Float16* qp = (_Float16*)d_ws;
  _Float16* kp = qp + (size_t)ROWS * DK;
  _Float16* vT = kp + (size_t)ROWS * DK;

  proj_relu_f16<<<2048, 128, 0, stream>>>(q, k, wq, wk, qp, kp);
  v_to_f16T<<<(ROWS * DK) / 256, 256, 0, stream>>>(v, vT);
  flash_attn<<<256, 128, 0, stream>>>(qp, kp, vT, out);
}
